// DevignModel_mod_62483184222418
// MI455X (gfx1250) — matrix operand order repaired and checked by reading, not compiled, not run on hardware
//
#include <hip/hip_runtime.h>

typedef __attribute__((ext_vector_type(16))) __bf16 v16bf;
typedef __attribute__((ext_vector_type(8)))  float  v8f;

#define TPB 256

__device__ __forceinline__ unsigned short f32_to_bf16(float f) {
  unsigned int u = __float_as_uint(f);
  u += 0x7FFFu + ((u >> 16) & 1u);        // round-to-nearest-even
  return (unsigned short)(u >> 16);
}
__device__ __forceinline__ float bf16_to_f32(unsigned short h) {
  return __uint_as_float((unsigned int)h << 16);
}

union ABfrag { v16bf v; uint4 q[2]; };

// C[M,N] = sum_s A_s[M,K] * B_s[K,N], 2x2 register-blocked per wave.
//   A: row-major bf16 [m][k], leading dim lda; shift s advances A by s*shiftA elems
//   B: pre-transposed bf16 Bt[n][k], leading dim ldb; shift s advances B by s*shiftB
// Each wave: M-tiles {2*mp, 2*mp+1} x N-tiles {2*bx, 2*bx+1}; Ntiles must be even.
// Second M-tile's loads stay in (zero-padded) bounds; its stores are guarded.
__global__ void k_gemm_wmma(const unsigned short* __restrict__ A,
                            const unsigned short* __restrict__ Bt,
                            void* __restrict__ Cout,
                            int Mtiles, int Kp,
                            int lda, int ldb, int ldc,
                            int shifts, int shiftA, long long shiftB,
                            int out_bf16)
{
  int wave  = threadIdx.x >> 5;
  int lane  = threadIdx.x & 31;
  int mpair = blockIdx.y * (TPB >> 5) + wave;
  int m0 = mpair * 2;
  if (m0 >= Mtiles) return;               // uniform per wave: EXEC stays all-ones
  const bool m1ok = (m0 + 1) < Mtiles;
  int n0 = blockIdx.x * 2;
  int hh  = lane >> 4;                    // half-wave select
  int l16 = lane & 15;

  v8f acc00 = {0.f,0.f,0.f,0.f,0.f,0.f,0.f,0.f};
  v8f acc01 = acc00, acc10 = acc00, acc11 = acc00;

  for (int s = 0; s < shifts; ++s) {
    const unsigned short* Ab0 = A  + (size_t)s * shiftA
                                   + (size_t)(m0 * 16 + l16) * lda + hh * 8;
    const unsigned short* Ab1 = Ab0 + (size_t)16 * lda;
    const unsigned short* Bb0 = Bt + (size_t)s * shiftB
                                   + (size_t)(n0 * 16 + l16) * ldb + hh * 8;
    const unsigned short* Bb1 = Bb0 + (size_t)16 * ldb;
    for (int kb = 0; kb < Kp; kb += 32) {
      // speculative stream prefetch ~256B ahead (global_prefetch_b8; OOB dropped)
      __builtin_prefetch(Ab0 + kb + 128, 0, 3);
      __builtin_prefetch(Bb0 + kb + 128, 0, 3);
      ABfrag fa0, fa1, fb0, fb1;
      // A 16-bit layout: lanes0-15 elems0-7 -> K 0-7, elems8-15 -> K 16-23; lanes16-31 +8
      fa0.q[0] = *(const uint4*)(Ab0 + kb);
      fa0.q[1] = *(const uint4*)(Ab0 + kb + 16);
      fa1.q[0] = *(const uint4*)(Ab1 + kb);
      fa1.q[1] = *(const uint4*)(Ab1 + kb + 16);
      // B layout: same K order as A (elems0-7 -> K 0-7, elems8-15 -> K 16-23; lanes16-31 +8), column = lane&15
      fb0.q[0] = *(const uint4*)(Bb0 + kb);
      fb0.q[1] = *(const uint4*)(Bb0 + kb + 16);
      fb1.q[0] = *(const uint4*)(Bb1 + kb);
      fb1.q[1] = *(const uint4*)(Bb1 + kb + 16);
      // 4 independent WMMAs: full operand reuse, no RAW hazards between them
      acc00 = __builtin_amdgcn_wmma_f32_16x16x32_bf16(false, fa0.v, false, fb0.v,
                                                      (short)0, acc00, false, false);
      acc01 = __builtin_amdgcn_wmma_f32_16x16x32_bf16(false, fa0.v, false, fb1.v,
                                                      (short)0, acc01, false, false);
      acc10 = __builtin_amdgcn_wmma_f32_16x16x32_bf16(false, fa1.v, false, fb0.v,
                                                      (short)0, acc10, false, false);
      acc11 = __builtin_amdgcn_wmma_f32_16x16x32_bf16(false, fa1.v, false, fb1.v,
                                                      (short)0, acc11, false, false);
    }
  }
  int col0 = n0 * 16 + l16;
  int col1 = col0 + 16;
  int row0 = m0 * 16 + hh * 8;            // C/D: VGPR r -> M = r + 8*half
  int row1 = row0 + 16;
  if (out_bf16) {
    unsigned short* C = (unsigned short*)Cout;
#pragma unroll
    for (int r = 0; r < 8; ++r) {
      C[(size_t)(row0 + r) * ldc + col0] = f32_to_bf16(acc00[r]);
      C[(size_t)(row0 + r) * ldc + col1] = f32_to_bf16(acc01[r]);
    }
    if (m1ok) {
#pragma unroll
      for (int r = 0; r < 8; ++r) {
        C[(size_t)(row1 + r) * ldc + col0] = f32_to_bf16(acc10[r]);
        C[(size_t)(row1 + r) * ldc + col1] = f32_to_bf16(acc11[r]);
      }
    }
  } else {
    float* C = (float*)Cout;
#pragma unroll
    for (int r = 0; r < 8; ++r) {
      C[(size_t)(row0 + r) * ldc + col0] = acc00[r];
      C[(size_t)(row0 + r) * ldc + col1] = acc01[r];
    }
    if (m1ok) {
#pragma unroll
      for (int r = 0; r < 8; ++r) {
        C[(size_t)(row1 + r) * ldc + col0] = acc10[r];
        C[(size_t)(row1 + r) * ldc + col1] = acc11[r];
      }
    }
  }
}

// ---- packing / elementwise kernels -----------------------------------------

__global__ void k_zero_f32(float* __restrict__ p, long long n) {
  long long i = (long long)blockIdx.x * TPB + threadIdx.x;
  if (i < n) p[i] = 0.f;
}

// f32 [rows][srcld] -> bf16 [rowsT][ldd], zero-padded
__global__ void k_pack_bf16(const float* __restrict__ src, unsigned short* __restrict__ dst,
                            int rows, int cols, int srcld, long long total, int ldd) {
  long long i = (long long)blockIdx.x * TPB + threadIdx.x;
  if (i >= total) return;
  int r = (int)(i / ldd), c = (int)(i % ldd);
  float v = (r < rows && c < cols) ? src[(size_t)r * srcld + c] : 0.f;
  dst[i] = f32_to_bf16(v);
}

// transpose-pack per layer: dst[l][j][k] = src[l][k][j]   (src [layers][K][J] f32)
__global__ void k_pack_wT(const float* __restrict__ src, unsigned short* __restrict__ dst,
                          int layers, int J, int K, int JP, int KP) {
  long long i = (long long)blockIdx.x * TPB + threadIdx.x;
  long long total = (long long)layers * JP * KP;
  if (i >= total) return;
  int l = (int)(i / ((long long)JP * KP));
  int rem = (int)(i % ((long long)JP * KP));
  int j = rem / KP, k = rem % KP;
  float v = (j < J && k < K) ? src[(size_t)l * J * K + (size_t)k * J + j] : 0.f;
  dst[i] = f32_to_bf16(v);
}

// pad-pack (no transpose): dst[j][k] = src[j][k]
__global__ void k_pack_nT(const float* __restrict__ src, unsigned short* __restrict__ dst,
                          int J, int K, int JP, int KP) {
  long long i = (long long)blockIdx.x * TPB + threadIdx.x;
  if (i >= (long long)JP * KP) return;
  int j = (int)(i / KP), k = (int)(i % KP);
  float v = (j < J && k < K) ? src[(size_t)j * K + k] : 0.f;
  dst[i] = f32_to_bf16(v);
}

// conv weight [O][I][S] f32 -> bf16 [S][OP][IP]
__global__ void k_pack_conv(const float* __restrict__ src, unsigned short* __restrict__ dst,
                            int O, int I, int S, int OP, int IP) {
  long long i = (long long)blockIdx.x * TPB + threadIdx.x;
  long long total = (long long)S * OP * IP;
  if (i >= total) return;
  int s = (int)(i / ((long long)OP * IP));
  int rem = (int)(i % ((long long)OP * IP));
  int o = rem / IP, c = rem % IP;
  float v = (o < O && c < I) ? src[((size_t)o * I + c) * S + s] : 0.f;
  dst[i] = f32_to_bf16(v);
}

// h0 = pad(x, OUT) in f32 [N][ldh]
__global__ void k_init_h(const float* __restrict__ x, float* __restrict__ h,
                         int N, int IN, int ldh) {
  long long i = (long long)blockIdx.x * TPB + threadIdx.x;
  if (i >= (long long)N * ldh) return;
  int r = (int)(i / ldh), c = (int)(i % ldh);
  h[i] = (c < IN) ? x[(size_t)r * IN + c] : 0.f;
}

// cb = concat(h, x) bf16 [rowsT][ldd]
__global__ void k_concat(const float* __restrict__ h, const float* __restrict__ x,
                         unsigned short* __restrict__ dst,
                         int N, int OUT, int IN, int ldh, long long total, int ldd) {
  long long i = (long long)blockIdx.x * TPB + threadIdx.x;
  if (i >= total) return;
  int r = (int)(i / ldd), c = (int)(i % ldd);
  float v = 0.f;
  if (r < N) {
    if (c < OUT)            v = h[(size_t)r * ldh + c];
    else if (c < OUT + IN)  v = x[(size_t)r * IN + (c - OUT)];
  }
  dst[i] = f32_to_bf16(v);
}

// agg[dst] += m[src]  (bf16 message, f32 atomic accumulate; 4 channels / thread)
__global__ void k_scatter(const unsigned short* __restrict__ m16,
                          const int* __restrict__ esrc, const int* __restrict__ edst,
                          float* __restrict__ agg, long long total, int groups, int ld) {
  long long i = (long long)blockIdx.x * TPB + threadIdx.x;
  if (i >= total) return;
  int e = (int)(i / groups);
  int g = (int)(i % groups);
  int s = esrc[e], d = edst[e];
  const unsigned short* mp = m16 + (size_t)s * ld + g * 4;
  float* ap = agg + (size_t)d * ld + g * 4;
  uint2 pk = *(const uint2*)mp;
  __hip_atomic_fetch_add(ap + 0, bf16_to_f32((unsigned short)(pk.x & 0xFFFFu)), __ATOMIC_RELAXED, __HIP_MEMORY_SCOPE_AGENT);
  __hip_atomic_fetch_add(ap + 1, bf16_to_f32((unsigned short)(pk.x >> 16)),     __ATOMIC_RELAXED, __HIP_MEMORY_SCOPE_AGENT);
  __hip_atomic_fetch_add(ap + 2, bf16_to_f32((unsigned short)(pk.y & 0xFFFFu)), __ATOMIC_RELAXED, __HIP_MEMORY_SCOPE_AGENT);
  __hip_atomic_fetch_add(ap + 3, bf16_to_f32((unsigned short)(pk.y >> 16)),     __ATOMIC_RELAXED, __HIP_MEMORY_SCOPE_AGENT);
}

// GRU gates: h = (1-z)*n + z*h
__global__ void k_gru_gate(const float* __restrict__ gi, const float* __restrict__ gh,
                           const float* __restrict__ bih, const float* __restrict__ bhh,
                           float* __restrict__ h, long long total, int C, int ldg, int ldh) {
  long long i = (long long)blockIdx.x * TPB + threadIdx.x;
  if (i >= total) return;
  int r = (int)(i / C), c = (int)(i % C);
  const float* gir = gi + (size_t)r * ldg;
  const float* ghr = gh + (size_t)r * ldg;
  float ir = gir[c]         + bih[c];
  float iz = gir[C + c]     + bih[C + c];
  float ic = gir[2 * C + c] + bih[2 * C + c];
  float hr = ghr[c]         + bhh[c];
  float hz = ghr[C + c]     + bhh[C + c];
  float hc = ghr[2 * C + c] + bhh[2 * C + c];
  float rg = 1.f / (1.f + __expf(-(ir + hr)));
  float zg = 1.f / (1.f + __expf(-(iz + hz)));
  float ng = tanhf(ic + rg * hc);
  float* hp = h + (size_t)r * ldh + c;
  *hp = (1.f - zg) * ng + zg * (*hp);
}

// BN training stats -> fused scale/shift per channel (one block / channel)
__global__ void k_bn_stats(const float* __restrict__ y, int L, int ldc,
                           const float* __restrict__ g, const float* __restrict__ b,
                           float* __restrict__ scale, float* __restrict__ shiftv) {
  int c = blockIdx.x;
  __shared__ float s1[TPB], s2[TPB];
  float a = 0.f, q = 0.f;
  for (int l = threadIdx.x; l < L; l += TPB) {
    float v = y[(size_t)l * ldc + c];
    a += v; q += v * v;
  }
  s1[threadIdx.x] = a; s2[threadIdx.x] = q;
  __syncthreads();
  for (int st = TPB / 2; st > 0; st >>= 1) {
    if (threadIdx.x < st) { s1[threadIdx.x] += s1[threadIdx.x + st]; s2[threadIdx.x] += s2[threadIdx.x + st]; }
    __syncthreads();
  }
  if (threadIdx.x == 0) {
    float mu  = s1[0] / (float)L;
    float var = s2[0] / (float)L - mu * mu;
    float sc  = g[c] * rsqrtf(var + 1e-5f);
    scale[c]  = sc;
    shiftv[c] = b[c] - mu * sc;
  }
}

// out[l'][c] = bf16( max(0, max_w bn(in[l'*s+w][c])) ); zero pad rows/cols
__global__ void k_bn_relu_pool(const float* __restrict__ y, unsigned short* __restrict__ out,
                               const float* __restrict__ scale, const float* __restrict__ shiftv,
                               int Lout, int Cvalid, int ldc_in, int ldd, long long total,
                               int k, int s) {
  long long i = (long long)blockIdx.x * TPB + threadIdx.x;
  if (i >= total) return;
  int r = (int)(i / ldd), c = (int)(i % ldd);
  unsigned short o = 0;
  if (r < Lout && c < Cvalid) {
    float sc = scale[c], sh = shiftv[c];
    float m = 0.f;                       // relu floor
    for (int w = 0; w < k; ++w) {
      float v = y[(size_t)(r * s + w) * ldc_in + c] * sc + sh;
      m = fmaxf(m, v);
    }
    o = f32_to_bf16(m);
  }
  out[i] = o;
}

// sum over l of (Y2@wy + by)*(Z2@wz + bz)  into acc[2]
__global__ void k_final(const unsigned short* __restrict__ Y2, const unsigned short* __restrict__ Z2,
                        const float* __restrict__ wy, const float* __restrict__ by,
                        const float* __restrict__ wz, const float* __restrict__ bz,
                        float* __restrict__ acc, int L, int Cy, int Cz, int ldy, int ldz) {
  int l = blockIdx.x * TPB + threadIdx.x;
  float b0 = 0.f, b1 = 0.f;
  if (l < L) {
    const unsigned short* yr = Y2 + (size_t)l * ldy;
    float y0 = by[0], y1 = by[1];
    for (int c = 0; c < Cy; ++c) { float v = bf16_to_f32(yr[c]); y0 += v * wy[c]; y1 += v * wy[Cy + c]; }
    const unsigned short* zr = Z2 + (size_t)l * ldz;
    float z0 = bz[0], z1 = bz[1];
    for (int c = 0; c < Cz; ++c) { float v = bf16_to_f32(zr[c]); z0 += v * wz[c]; z1 += v * wz[Cz + c]; }
    b0 = y0 * z0; b1 = y1 * z1;
  }
  __shared__ float s0[TPB], s1[TPB];
  s0[threadIdx.x] = b0; s1[threadIdx.x] = b1;
  __syncthreads();
  for (int st = TPB / 2; st > 0; st >>= 1) {
    if (threadIdx.x < st) { s0[threadIdx.x] += s0[threadIdx.x + st]; s1[threadIdx.x] += s1[threadIdx.x + st]; }
    __syncthreads();
  }
  if (threadIdx.x == 0) {
    __hip_atomic_fetch_add(acc + 0, s0[0], __ATOMIC_RELAXED, __HIP_MEMORY_SCOPE_AGENT);
    __hip_atomic_fetch_add(acc + 1, s1[0], __ATOMIC_RELAXED, __HIP_MEMORY_SCOPE_AGENT);
  }
}

__global__ void k_softmax2(const float* __restrict__ acc, float* __restrict__ out, float invL) {
  if (threadIdx.x == 0) {
    float a0 = acc[0] * invL, a1 = acc[1] * invL;
    float m  = fmaxf(a0, a1);
    float e0 = expf(a0 - m), e1 = expf(a1 - m);
    float d  = e0 + e1;
    out[0] = e0 / d; out[1] = e1 / d;
  }
}

// ---- host orchestration -----------------------------------------------------

static inline int GB(long long n) { return (int)((n + TPB - 1) / TPB); }
static inline dim3 gemm_grid(int Ntiles, int Mtiles) {
  int mpairs = (Mtiles + 1) / 2;
  return dim3(Ntiles / 2, (mpairs + 7) / 8);
}

extern "C" void kernel_launch(void* const* d_in, const int* in_sizes, int n_in,
                              void* d_out, int out_size, void* d_ws, size_t ws_size,
                              hipStream_t stream) {
  (void)n_in; (void)out_size; (void)ws_size;
  const int IN_F = 100, OUT_F = 200, CAT_F = 300, STEPS = 8;
  const int OUTP = 224, CATP = 320, G3P = 608;       // padded dims (mult of 32)
  const int N = in_sizes[0] / IN_F;
  const int E = in_sizes[1] / 2;

  const float* x      = (const float*)d_in[0];
  const int*   esrc   = (const int*)d_in[1];
  const int*   edst   = esrc + E;
  const float* ggnn_w = (const float*)d_in[2];
  const float* wih    = (const float*)d_in[3];
  const float* whh    = (const float*)d_in[4];
  const float* bih    = (const float*)d_in[5];
  const float* bhh    = (const float*)d_in[6];
  const float* c1w    = (const float*)d_in[7];
  const float* c2w    = (const float*)d_in[9];
  const float* c1cw   = (const float*)d_in[11];
  const float* c2cw   = (const float*)d_in[13];
  const float* bng    = (const float*)d_in[15];
  const float* bnb    = (const float*)d_in[16];
  const float* bncg   = (const float*)d_in[17];
  const float* bncb   = (const float*)d_in[18];
  const float* mlpyw  = (const float*)d_in[19];
  const float* mlpyb  = (const float*)d_in[20];
  const float* mlpzw  = (const float*)d_in[21];
  const float* mlpzb  = (const float*)d_in[22];
  float* out = (float*)d_out;

  const int MtN  = (N + 15) / 16;             // 3125
  const int R_HB = MtN * 16 + 32;             // bf16 node rows (+2x2-block & shift overrun)
  const int R_A16 = ((MtN + 1) & ~1) * 16;    // agg bf16 rows, even #tiles
  const int L1   = N - 3;                     // conv1 output length
  const int Mt1  = (L1 + 15) / 16;
  const int R_C1 = Mt1 * 16;
  const int P1L  = (L1 - 4) / 2 + 1;          // pool1 length
  const int C2L  = P1L - 1;                   // conv2 output length
  const int Mt2  = (C2L + 15) / 16;
  const int R_P1 = Mt2 * 16 + 32;
  const int R_C2 = Mt2 * 16;
  const int P2L  = (C2L - 3) / 2 + 1;         // pool2 length (both branches)
  const int R_P2 = ((P2L + 15) / 16) * 16;
  const int CHUNK = 10000;                    // GRU row chunk (multiple of 16 rows)

  // ---- workspace layout (phase A GGNN scratch overlaid by phase B conv scratch)
  char* ws = (char*)d_ws;
  size_t off = 0;
  auto take = [&](size_t b) -> char* {
    char* p = ws + off;
    off += (b + 255) & ~(size_t)255;
    return p;
  };
  float*          H    = (float*)take((size_t)N * OUTP * 4);
  unsigned short* HB   = (unsigned short*)take((size_t)R_HB * OUTP * 2);
  size_t scr0 = off;
  unsigned short* M16   = (unsigned short*)take((size_t)N * OUTP * 2);
  float*          AGG   = (float*)take((size_t)N * OUTP * 4);
  unsigned short* AGG16 = (unsigned short*)take((size_t)R_A16 * OUTP * 2);
  float*          GI    = (float*)take((size_t)CHUNK * G3P * 4);
  float*          GH    = (float*)take((size_t)CHUNK * G3P * 4);
  size_t endA = off;
  off = scr0;                                  // overlay
  unsigned short* CB  = (unsigned short*)take((size_t)R_HB * CATP * 2);
  float*          C1O = (float*)take((size_t)R_C1 * CATP * 4);
  unsigned short* P1  = (unsigned short*)take((size_t)R_P1 * CATP * 2);
  float*          C2O = (float*)take((size_t)R_C2 * CATP * 4);
  size_t endB = off;
  off = (endA > endB) ? endA : endB;
  unsigned short* P2Y  = (unsigned short*)take((size_t)R_P2 * OUTP * 2);
  unsigned short* P2Z  = (unsigned short*)take((size_t)R_P2 * CATP * 2);
  unsigned short* WG   = (unsigned short*)take((size_t)STEPS * OUTP * OUTP * 2);
  unsigned short* WIH  = (unsigned short*)take((size_t)G3P * OUTP * 2);
  unsigned short* WHH  = (unsigned short*)take((size_t)G3P * OUTP * 2);
  unsigned short* WC1  = (unsigned short*)take((size_t)4 * OUTP * OUTP * 2);
  unsigned short* WC2  = (unsigned short*)take((size_t)2 * OUTP * OUTP * 2);
  unsigned short* WC1C = (unsigned short*)take((size_t)4 * CATP * CATP * 2);
  unsigned short* WC2C = (unsigned short*)take((size_t)2 * CATP * CATP * 2);
  float*          SCL  = (float*)take(CATP * 4);
  float*          SHF  = (float*)take(CATP * 4);
  float*          ACC2 = (float*)take(2 * 4);

  // ---- weight packing (bf16, padded, B-transposed layouts) ----
  k_pack_wT  <<<GB((long long)STEPS * OUTP * OUTP), TPB, 0, stream>>>(ggnn_w, WG, STEPS, OUT_F, OUT_F, OUTP, OUTP);
  k_pack_nT  <<<GB((long long)G3P * OUTP), TPB, 0, stream>>>(wih, WIH, 3 * OUT_F, OUT_F, G3P, OUTP);
  k_pack_nT  <<<GB((long long)G3P * OUTP), TPB, 0, stream>>>(whh, WHH, 3 * OUT_F, OUT_F, G3P, OUTP);
  k_pack_conv<<<GB((long long)4 * OUTP * OUTP), TPB, 0, stream>>>(c1w,  WC1,  OUT_F, OUT_F, 4, OUTP, OUTP);
  k_pack_conv<<<GB((long long)2 * OUTP * OUTP), TPB, 0, stream>>>(c2w,  WC2,  OUT_F, OUT_F, 2, OUTP, OUTP);
  k_pack_conv<<<GB((long long)4 * CATP * CATP), TPB, 0, stream>>>(c1cw, WC1C, CAT_F, CAT_F, 4, CATP, CATP);
  k_pack_conv<<<GB((long long)2 * CATP * CATP), TPB, 0, stream>>>(c2cw, WC2C, CAT_F, CAT_F, 2, CATP, CATP);

  // ---- GGNN: h0 = pad(x); 8 steps of (m = hW; agg = scatter; h = GRU(agg, h)) ----
  k_init_h<<<GB((long long)N * OUTP), TPB, 0, stream>>>(x, H, N, IN_F, OUTP);

  for (int st = 0; st < STEPS; ++st) {
    k_pack_bf16<<<GB((long long)R_HB * OUTP), TPB, 0, stream>>>(H, HB, N, OUTP, OUTP, (long long)R_HB * OUTP, OUTP);
    // m = h @ W_l   -> bf16 [N][OUTP]
    k_gemm_wmma<<<gemm_grid(OUTP / 16, MtN), TPB, 0, stream>>>(
        HB, WG + (size_t)st * OUTP * OUTP, M16, MtN, OUTP, OUTP, OUTP, OUTP, 1, 0, 0LL, 1);
    k_zero_f32<<<GB((long long)N * OUTP), TPB, 0, stream>>>(AGG, (long long)N * OUTP);
    k_scatter<<<GB((long long)E * (OUTP / 4)), TPB, 0, stream>>>(
        M16, esrc, edst, AGG, (long long)E * (OUTP / 4), OUTP / 4, OUTP);
    k_pack_bf16<<<GB((long long)R_A16 * OUTP), TPB, 0, stream>>>(AGG, AGG16, N, OUTP, OUTP, (long long)R_A16 * OUTP, OUTP);
    for (int n0 = 0; n0 < N; n0 += CHUNK) {
      int rows = (N - n0 < CHUNK) ? (N - n0) : CHUNK;
      int mt = (rows + 15) / 16;
      // gi = agg @ Wih^T ; gh = h @ Whh^T
      k_gemm_wmma<<<gemm_grid(G3P / 16, mt), TPB, 0, stream>>>(
          AGG16 + (size_t)n0 * OUTP, WIH, GI, mt, OUTP, OUTP, OUTP, G3P, 1, 0, 0LL, 0);
      k_gemm_wmma<<<gemm_grid(G3P / 16, mt), TPB, 0, stream>>>(
          HB + (size_t)n0 * OUTP, WHH, GH, mt, OUTP, OUTP, OUTP, G3P, 1, 0, 0LL, 0);
      k_gru_gate<<<GB((long long)rows * OUT_F), TPB, 0, stream>>>(
          GI, GH, bih, bhh, H + (size_t)n0 * OUTP, (long long)rows * OUT_F, OUT_F, G3P, OUTP);
    }
  }

  // ---- conv branches (conv1d as shifted accumulating GEMMs; bias absorbed by BN) ----
  k_pack_bf16<<<GB((long long)R_HB * OUTP), TPB, 0, stream>>>(H, HB, N, OUTP, OUTP, (long long)R_HB * OUTP, OUTP);
  k_concat<<<GB((long long)R_HB * CATP), TPB, 0, stream>>>(H, x, CB, N, OUT_F, IN_F, OUTP, (long long)R_HB * CATP, CATP);

  // Y branch (C = OUT)
  k_gemm_wmma<<<gemm_grid(OUTP / 16, Mt1), TPB, 0, stream>>>(
      HB, WC1, C1O, Mt1, OUTP, OUTP, OUTP, OUTP, 4, OUTP, (long long)OUTP * OUTP, 0);
  k_bn_stats<<<OUT_F, TPB, 0, stream>>>(C1O, L1, OUTP, bng, bnb, SCL, SHF);
  k_bn_relu_pool<<<GB((long long)R_P1 * OUTP), TPB, 0, stream>>>(
      C1O, P1, SCL, SHF, P1L, OUT_F, OUTP, OUTP, (long long)R_P1 * OUTP, 4, 2);
  k_gemm_wmma<<<gemm_grid(OUTP / 16, Mt2), TPB, 0, stream>>>(
      P1, WC2, C2O, Mt2, OUTP, OUTP, OUTP, OUTP, 2, OUTP, (long long)OUTP * OUTP, 0);
  k_bn_stats<<<OUT_F, TPB, 0, stream>>>(C2O, C2L, OUTP, bng, bnb, SCL, SHF);
  k_bn_relu_pool<<<GB((long long)R_P2 * OUTP), TPB, 0, stream>>>(
      C2O, P2Y, SCL, SHF, P2L, OUT_F, OUTP, OUTP, (long long)R_P2 * OUTP, 3, 2);

  // Z branch (C = CAT)
  k_gemm_wmma<<<gemm_grid(CATP / 16, Mt1), TPB, 0, stream>>>(
      CB, WC1C, C1O, Mt1, CATP, CATP, CATP, CATP, 4, CATP, (long long)CATP * CATP, 0);
  k_bn_stats<<<CAT_F, TPB, 0, stream>>>(C1O, L1, CATP, bncg, bncb, SCL, SHF);
  k_bn_relu_pool<<<GB((long long)R_P1 * CATP), TPB, 0, stream>>>(
      C1O, P1, SCL, SHF, P1L, CAT_F, CATP, CATP, (long long)R_P1 * CATP, 4, 2);
  k_gemm_wmma<<<gemm_grid(CATP / 16, Mt2), TPB, 0, stream>>>(
      P1, WC2C, C2O, Mt2, CATP, CATP, CATP, CATP, 2, CATP, (long long)CATP * CATP, 0);
  k_bn_stats<<<CAT_F, TPB, 0, stream>>>(C2O, C2L, CATP, bncg, bncb, SCL, SHF);
  k_bn_relu_pool<<<GB((long long)R_P2 * CATP), TPB, 0, stream>>>(
      C2O, P2Z, SCL, SHF, P2L, CAT_F, CATP, CATP, (long long)R_P2 * CATP, 3, 2);

  // ---- final MLP product, mean over L, softmax ----
  k_zero_f32<<<1, TPB, 0, stream>>>(ACC2, 2);
  k_final<<<GB(P2L), TPB, 0, stream>>>(P2Y, P2Z, mlpyw, mlpyb, mlpzw, mlpzb,
                                       ACC2, P2L, OUT_F, CAT_F, OUTP, CATP);
  k_softmax2<<<1, 32, 0, stream>>>(ACC2, out, 1.f / (float)P2L);
}
